// GeneralE3GNN2_74629351735701
// MI455X (gfx1250) — matrix operand order repaired and checked by reading, not compiled, not run on hardware
//
#include <hip/hip_runtime.h>
#include <hip/hip_bf16.h>

// ---------------- problem constants (match reference) ----------------
#define H      128
#define Bg     32
#define Lg     256
#define Ng     (Bg * Lg)      // 8192 nodes
#define KNN    30
#define Eg     (Ng * KNN)     // 245760 edges
#define NRBF   32
#define NHEADS 4
#define DHEAD  (H / NHEADS)   // 32

typedef __attribute__((ext_vector_type(16))) _Float16 v16h;
typedef __attribute__((ext_vector_type(4)))  _Float16 v4h;
typedef __attribute__((ext_vector_type(8)))  float    v8f;

// ---------------- activation ----------------
__device__ __forceinline__ float activate(float x, int act) {
  switch (act) {
    case 1: return x > 0.f ? x : 0.f;                                 // relu
    case 2: return 0.5f * x * (1.f + erff(x * 0.70710678118654752f)); // exact gelu
    case 3: return 1.f / (1.f + __expf(-x));                          // sigmoid
  }
  return x;
}

// ---------------- WMMA GEMM common geometry ----------------
// 256 threads = 8 waves. Block tile 128(M) x 32(N), K stepped by 32.
// Tiles staged into LDS in *fragment order* so each lane reads its whole
// 16-half WMMA operand as one contiguous 32-byte LDS access (ds_load_b128 x2).
// Mapping (CDNA5 ISA 16-bit operand layouts):
//   A 16x32 (MxK): lane = (m&15) + 16*((k>>3)&1), slot = (k&7) + 8*((k>>4)&1)
//   B 32x16 (KxN): lane = (n&15) + 16*((k>>3)&1), slot = (k&7) + 8*((k>>4)&1)
//   C f32 16x16:   row = v + 8*(lane>>4), col = lane&15
// Each wave: one A fragment x two B sub-tiles -> 2 WMMAs per K step.
#define BT_M 128
#define BT_N 32
#define BT_K 32

// ======== fully tile-aligned fast path: M%128==0, Kd%32==0, Nn%32==0 ========
// Double-buffered LDS, one barrier per K step, software-pipelined staging:
//   frag ds_loads -> issue next tile's global loads -> WMMAs -> cvt+ds_store.
__global__ __launch_bounds__(256) void gemm_full_kernel(
    const float* __restrict__ A, const float* __restrict__ W,
    const float* __restrict__ bias, float* __restrict__ C,
    int M, int Kd, int Nn, int act)
{
  __shared__ __align__(32) _Float16 Af[2][8][32][16];  // 16 KB (buf stride 4096 halves)
  __shared__ __align__(32) _Float16 Bf[2][2][32][16];  //  4 KB (buf stride 1024 halves)

  const int tid  = threadIdx.x;
  const int lane = tid & 31;
  const int wave = tid >> 5;
  const int m0   = blockIdx.y * BT_M;
  const int n0   = blockIdx.x * BT_N;

  // ---- constant per-thread staging geometry ----
  // A: thread covers rows rbase + 32*it (it=0..3), columns c4..c4+3 of the K tile.
  const int c4    = (tid & 7) << 2;
  const int hlA   = (c4 >> 3) & 1;
  const int slotA = (c4 & 7) + (((c4 >> 4) & 1) << 3);
  const int rbase = tid >> 3;                   // 0..31
  const float*  pA0    = &A[(size_t)(m0 + rbase) * Kd + c4];
  _Float16*     sAbase = &Af[0][rbase >> 4][(rbase & 15) + 16 * hlA][slotA]; // +it*1024, +buf*4096
  // B: thread covers row kB, columns n4..n4+3.
  const int kB    = tid >> 3;                   // 0..31
  const int n4    = (tid & 7) << 2;
  const int slotB = (kB & 7) + (((kB >> 4) & 1) << 3);
  const int blB   = (n4 & 15) + 16 * ((kB >> 3) & 1);
  const float*  pB0    = &W[(size_t)kB * Nn + n0 + n4];
  _Float16*     sBbase = &Bf[0][n4 >> 4][blB][slotB];                        // +buf*1024, col j -> +j*16
  // fragment read bases
  const _Float16* fragA = &Af[0][wave][lane][0];
  const _Float16* fragB = &Bf[0][0][lane][0];   // b1 at +512 halves

  float4 fa0, fa1, fa2, fa3, fb;
  auto loadTiles = [&](int k0) {
    fa0 = *(const float4*)(pA0 + (size_t)k0);
    fa1 = *(const float4*)(pA0 + (size_t)32 * Kd + k0);
    fa2 = *(const float4*)(pA0 + (size_t)64 * Kd + k0);
    fa3 = *(const float4*)(pA0 + (size_t)96 * Kd + k0);
    fb  = *(const float4*)(pB0 + (size_t)k0 * Nn);
  };
  auto storeTiles = [&](int buf) {
    _Float16* sa = (_Float16*)sAbase + buf * 4096;
    *(v4h*)(sa +    0) = (v4h){(_Float16)fa0.x, (_Float16)fa0.y, (_Float16)fa0.z, (_Float16)fa0.w};
    *(v4h*)(sa + 1024) = (v4h){(_Float16)fa1.x, (_Float16)fa1.y, (_Float16)fa1.z, (_Float16)fa1.w};
    *(v4h*)(sa + 2048) = (v4h){(_Float16)fa2.x, (_Float16)fa2.y, (_Float16)fa2.z, (_Float16)fa2.w};
    *(v4h*)(sa + 3072) = (v4h){(_Float16)fa3.x, (_Float16)fa3.y, (_Float16)fa3.z, (_Float16)fa3.w};
    _Float16* sb = (_Float16*)sBbase + buf * 1024;
    sb[ 0] = (_Float16)fb.x;
    sb[16] = (_Float16)fb.y;
    sb[32] = (_Float16)fb.z;
    sb[48] = (_Float16)fb.w;
  };

  v8f acc0 = {}, acc1 = {};
  const int nk = Kd / BT_K;

  loadTiles(0);
  storeTiles(0);
  __syncthreads();

  for (int ki = 0; ki < nk; ++ki) {
    const int buf = ki & 1;
    v16h af = *(const v16h*)(fragA + buf * 4096);   // ds_load_b128 x2
    v16h b0 = *(const v16h*)(fragB + buf * 1024);
    v16h b1 = *(const v16h*)(fragB + buf * 1024 + 512);
    const bool more = (ki + 1) < nk;
    if (more) loadTiles((ki + 1) * BT_K);           // global loads in flight over WMMAs
    if (ki + 2 < nk)                                // global_prefetch_b8 for K+2 tile
      __builtin_prefetch(pA0 + (size_t)(ki + 2) * BT_K, 0, 1);
    acc0 = __builtin_amdgcn_wmma_f32_16x16x32_f16(false, af, false, b0, (short)0, acc0, false, false);
    acc1 = __builtin_amdgcn_wmma_f32_16x16x32_f16(false, af, false, b1, (short)0, acc1, false, false);
    if (more) storeTiles(buf ^ 1);                  // cvt + ds_store after WMMAs
    __syncthreads();
  }

  const int hl = lane >> 4, l15 = lane & 15;
  #pragma unroll
  for (int fv = 0; fv < 8; ++fv) {
    int gm  = m0 + wave * 16 + fv + 8 * hl;
    int gn0 = n0 + l15;
    int gn1 = n0 + 16 + l15;
    float x0 = acc0[fv], x1 = acc1[fv];
    if (bias) { x0 += bias[gn0]; x1 += bias[gn1]; }
    C[(size_t)gm * Nn + gn0] = activate(x0, act);
    C[(size_t)gm * Nn + gn1] = activate(x1, act);
  }
}

// ======== generic guarded path (ragged dims: K=12, N=4, N=12, M=32) ========
__global__ __launch_bounds__(256) void gemm_gen_kernel(
    const float* __restrict__ A, const float* __restrict__ W,
    const float* __restrict__ bias, float* __restrict__ C,
    int M, int Kd, int Nn, int act)
{
  __shared__ __align__(32) _Float16 Af[8][32][16];  // 8 KB
  __shared__ __align__(32) _Float16 Bf[2][32][16];  // 2 KB

  const int tid  = threadIdx.x;
  const int lane = tid & 31;
  const int wave = tid >> 5;
  const int m0   = blockIdx.y * BT_M;
  const int n0   = blockIdx.x * BT_N;

  v8f acc0 = {}, acc1 = {};

  for (int k0 = 0; k0 < Kd; k0 += BT_K) {
    for (int t = tid; t < 1024; t += 256) {
      int r  = t >> 3;
      int c4 = (t & 7) << 2;
      int gm = m0 + r;
      v4h h;
      if (gm < M && (k0 + c4 + 4) <= Kd) {
        const float4 f = *(const float4*)&A[(size_t)gm * Kd + k0 + c4];
        h = (v4h){(_Float16)f.x, (_Float16)f.y, (_Float16)f.z, (_Float16)f.w};
      } else {
        float e0 = (gm < M && (k0 + c4 + 0) < Kd) ? A[(size_t)gm * Kd + k0 + c4 + 0] : 0.f;
        float e1 = (gm < M && (k0 + c4 + 1) < Kd) ? A[(size_t)gm * Kd + k0 + c4 + 1] : 0.f;
        float e2 = (gm < M && (k0 + c4 + 2) < Kd) ? A[(size_t)gm * Kd + k0 + c4 + 2] : 0.f;
        float e3 = (gm < M && (k0 + c4 + 3) < Kd) ? A[(size_t)gm * Kd + k0 + c4 + 3] : 0.f;
        h = (v4h){(_Float16)e0, (_Float16)e1, (_Float16)e2, (_Float16)e3};
      }
      int hl   = (c4 >> 3) & 1;
      int slot = (c4 & 7) + (((c4 >> 4) & 1) << 3);
      *(v4h*)&Af[r >> 4][(r & 15) + 16 * hl][slot] = h;
    }
    for (int t = tid; t < 1024; t += 256) {
      int n = t & 31, k = t >> 5;
      int gk = k0 + k, gn = n0 + n;
      float x = (gk < Kd && gn < Nn) ? W[(size_t)gk * Nn + gn] : 0.f;
      Bf[n >> 4][(n & 15) + 16 * ((k >> 3) & 1)][(k & 7) + (((k >> 4) & 1) << 3)] = (_Float16)x;
    }
    __syncthreads();

    v16h af = *(const v16h*)&Af[wave][lane][0];
    v16h b0 = *(const v16h*)&Bf[0][lane][0];
    v16h b1 = *(const v16h*)&Bf[1][lane][0];
    acc0 = __builtin_amdgcn_wmma_f32_16x16x32_f16(false, af, false, b0, (short)0, acc0, false, false);
    acc1 = __builtin_amdgcn_wmma_f32_16x16x32_f16(false, af, false, b1, (short)0, acc1, false, false);
    __syncthreads();
  }

  const int hl = lane >> 4, l15 = lane & 15;
  #pragma unroll
  for (int fv = 0; fv < 8; ++fv) {
    int gm = m0 + wave * 16 + fv + 8 * hl;
    if (gm >= M) continue;
    int gn0 = n0 + l15;
    if (gn0 < Nn) {
      float x = acc0[fv];
      if (bias) x += bias[gn0];
      C[(size_t)gm * Nn + gn0] = activate(x, act);
    }
    int gn1 = n0 + 16 + l15;
    if (gn1 < Nn) {
      float x = acc1[fv];
      if (bias) x += bias[gn1];
      C[(size_t)gm * Nn + gn1] = activate(x, act);
    }
  }
}

// ---------------- kNN (per-graph, CA atom = X[:,1,:], no self loops) ----------------
__global__ void knn_kernel(const float* __restrict__ X, int* __restrict__ dst) {
  int i = blockIdx.x * blockDim.x + threadIdx.x;
  if (i >= Ng) return;
  int b = i / Lg, il = i % Lg;
  float xi = X[(size_t)i * 12 + 3], yi = X[(size_t)i * 12 + 4], zi = X[(size_t)i * 12 + 5];
  float bd[KNN]; int bidx[KNN];
  #pragma unroll
  for (int k = 0; k < KNN; ++k) { bd[k] = 3.0e38f; bidx[k] = b * Lg; }
  for (int j = 0; j < Lg; ++j) {
    if (j == il) continue;
    int gj = b * Lg + j;
    float dx = X[(size_t)gj * 12 + 3] - xi;
    float dy = X[(size_t)gj * 12 + 4] - yi;
    float dz = X[(size_t)gj * 12 + 5] - zi;
    float d2 = dx * dx + dy * dy + dz * dz;
    if (d2 < bd[KNN - 1]) {
      int p = KNN - 1;
      while (p > 0 && bd[p - 1] > d2) { bd[p] = bd[p - 1]; bidx[p] = bidx[p - 1]; --p; }
      bd[p] = d2; bidx[p] = gj;
    }
  }
  for (int k = 0; k < KNN; ++k) dst[i * KNN + k] = bidx[k];
}

// ---------------- gathers / concats ----------------
__global__ void cat2_kernel(const float* __restrict__ hV, const int* __restrict__ dst,
                            float* __restrict__ out) {
  long t = blockIdx.x * (long)blockDim.x + threadIdx.x;
  if (t >= (long)Eg * 2 * H) return;
  int e = (int)(t / (2 * H)), c = (int)(t % (2 * H));
  int s = e / KNN;
  out[t] = (c < H) ? hV[(size_t)s * H + c] : hV[(size_t)dst[e] * H + (c - H)];
}

__global__ void cat3_kernel(const float* __restrict__ hV, const float* __restrict__ hE,
                            const int* __restrict__ dst, float* __restrict__ out) {
  long t = blockIdx.x * (long)blockDim.x + threadIdx.x;
  if (t >= (long)Eg * 3 * H) return;
  int e = (int)(t / (3 * H)), c = (int)(t % (3 * H));
  float x;
  if (c < H)          x = hV[(size_t)(e / KNN) * H + c];
  else if (c < 2 * H) x = hE[(size_t)e * H + (c - H)];
  else                x = hV[(size_t)dst[e] * H + (c - 2 * H)];
  out[t] = x;
}

// ---------------- RBF features [E, NRBF] ----------------
__global__ void rbf_kernel(const float* __restrict__ nidx, const int* __restrict__ dst,
                           float* __restrict__ out) {
  long t = blockIdx.x * (long)blockDim.x + threadIdx.x;
  if (t >= (long)Eg * NRBF) return;
  int e = (int)(t / NRBF), r = (int)(t % NRBF);
  float D  = nidx[e / KNN] - nidx[dst[e]];
  float mu = 20.f * (float)r / (float)(NRBF - 1);
  float z  = (D - mu) * ((float)NRBF / 20.f);
  out[t] = __expf(-z * z);
}

// ---------------- chain embedding add ----------------
__global__ void chain_add_kernel(float* __restrict__ hE, const int* __restrict__ chain,
                                 const int* __restrict__ dst, const float* __restrict__ cemb) {
  long t = blockIdx.x * (long)blockDim.x + threadIdx.x;
  if (t >= (long)Eg * H) return;
  int e = (int)(t / H), c = (int)(t % H);
  int same = (chain[e / KNN] == chain[dst[e]]) ? 1 : 0;
  hE[t] += cemb[same * H + c];
}

// ---------------- training-mode BatchNorm (two pass) ----------------
__global__ __launch_bounds__(256) void bn_stats_kernel(
    const float* __restrict__ X, int M, int C, float* __restrict__ mean, float* __restrict__ var) {
  __shared__ float ssum[256], ssq[256];
  int c = blockIdx.x;
  float s = 0.f, q = 0.f;
  for (int r = threadIdx.x; r < M; r += 256) {
    float x = X[(size_t)r * C + c];
    s += x; q += x * x;
  }
  ssum[threadIdx.x] = s; ssq[threadIdx.x] = q;
  __syncthreads();
  for (int o = 128; o > 0; o >>= 1) {
    if (threadIdx.x < o) { ssum[threadIdx.x] += ssum[threadIdx.x + o]; ssq[threadIdx.x] += ssq[threadIdx.x + o]; }
    __syncthreads();
  }
  if (threadIdx.x == 0) {
    float m = ssum[0] / (float)M;
    mean[c] = m;
    var[c]  = ssq[0] / (float)M - m * m;   // biased variance
  }
}

__global__ void bn_apply_kernel(const float* __restrict__ X, float* __restrict__ Y,
                                const float* __restrict__ mean, const float* __restrict__ var,
                                const float* __restrict__ g, const float* __restrict__ b,
                                long M, int C, int act) {
  long t = blockIdx.x * (long)blockDim.x + threadIdx.x;
  if (t >= M * (long)C) return;
  int c = (int)(t % C);
  float x = (X[t] - mean[c]) * rsqrtf(var[c] + 1e-5f) * g[c] + b[c];
  Y[t] = activate(x, act);
}

// ---------------- segment softmax over contiguous 30-edge runs ----------------
__global__ void segsoftmax_kernel(const float* __restrict__ w, float* __restrict__ att) {
  int t = blockIdx.x * blockDim.x + threadIdx.x;
  if (t >= Ng * NHEADS) return;
  int i = t / NHEADS, h = t % NHEADS;
  const float sc = 0.17677669529663687f;   // 1/sqrt(32)
  float mx = -3.0e38f;
  for (int k = 0; k < KNN; ++k)
    mx = fmaxf(mx, w[(size_t)(i * KNN + k) * NHEADS + h] * sc);
  float s = 0.f;
  for (int k = 0; k < KNN; ++k)
    s += __expf(w[(size_t)(i * KNN + k) * NHEADS + h] * sc - mx);
  float inv = 1.f / s;
  for (int k = 0; k < KNN; ++k) {
    size_t e = (size_t)(i * KNN + k) * NHEADS + h;
    att[e] = __expf(w[e] * sc - mx) * inv;
  }
}

// ---------------- agg[i,c] = sum_k att[e,h] * V[e,c] ----------------
__global__ void agg_kernel(const float* __restrict__ att, const float* __restrict__ V,
                           float* __restrict__ agg) {
  int t = blockIdx.x * blockDim.x + threadIdx.x;
  if (t >= Ng * H) return;
  int i = t / H, c = t % H, h = c / DHEAD;
  float s = 0.f;
  for (int k = 0; k < KNN; ++k) {
    int e = i * KNN + k;
    s += att[(size_t)e * NHEADS + h] * V[(size_t)e * H + c];
  }
  agg[t] = s;
}

// ---------------- elementwise helpers ----------------
__global__ void add_kernel(float* __restrict__ A, const float* __restrict__ Bv, long n) {
  long t = blockIdx.x * (long)blockDim.x + threadIdx.x;
  if (t < n) A[t] += Bv[t];
}
__global__ void add2_kernel(float* __restrict__ O, const float* __restrict__ A,
                            const float* __restrict__ Bv, long n) {
  long t = blockIdx.x * (long)blockDim.x + threadIdx.x;
  if (t < n) O[t] = A[t] + Bv[t];
}
__global__ void copy_kernel(float* __restrict__ O, const float* __restrict__ A, long n) {
  long t = blockIdx.x * (long)blockDim.x + threadIdx.x;
  if (t < n) O[t] = A[t];
}
__global__ void gated_res_kernel(float* __restrict__ hV, const float* __restrict__ t2,
                                 const float* __restrict__ g, long n) {
  long t = blockIdx.x * (long)blockDim.x + threadIdx.x;
  if (t < n) hV[t] += t2[t] * (1.f / (1.f + __expf(-g[t])));
}
__global__ void graph_mean_kernel(const float* __restrict__ hV, float* __restrict__ cv) {
  int t = blockIdx.x * blockDim.x + threadIdx.x;
  if (t >= Bg * H) return;
  int b = t / H, c = t % H;
  float s = 0.f;
  for (int r = 0; r < Lg; ++r) s += hV[(size_t)(b * Lg + r) * H + c];
  cv[t] = s * (1.f / (float)Lg);
}
__global__ void graph_gate_kernel(float* __restrict__ hV, const float* __restrict__ g) {
  long t = blockIdx.x * (long)blockDim.x + threadIdx.x;
  if (t >= (long)Ng * H) return;
  int i = (int)(t / H), c = (int)(t % H);
  hV[t] *= 1.f / (1.f + __expf(-g[(i / Lg) * H + c]));
}

// ---------------- host-side orchestration ----------------
static inline void launch_gemm(const float* A, const float* W, const float* bias, float* C,
                               int M, int Kd, int Nn, int act, hipStream_t s) {
  dim3 grid((Nn + BT_N - 1) / BT_N, (M + BT_M - 1) / BT_M);
  if ((M % BT_M) == 0 && (Kd % BT_K) == 0 && (Nn % BT_N) == 0)
    gemm_full_kernel<<<grid, dim3(256), 0, s>>>(A, W, bias, C, M, Kd, Nn, act);
  else
    gemm_gen_kernel<<<grid, dim3(256), 0, s>>>(A, W, bias, C, M, Kd, Nn, act);
}
static inline dim3 ew_grid(long n) { return dim3((unsigned)((n + 255) / 256)); }

extern "C" void kernel_launch(void* const* d_in, const int* in_sizes, int n_in,
                              void* d_out, int out_size, void* d_ws, size_t ws_size,
                              hipStream_t stream) {
  (void)in_sizes; (void)n_in; (void)out_size; (void)ws_size;

  // ---- inputs (setup_inputs order) ----
  const float* Xin   = (const float*)d_in[0];   // [N,4,3]
  const float* nidx  = (const float*)d_in[1];   // [N]
  const float* hV0   = (const float*)d_in[2];   // [N,H]
  const int*   chain = (const int*)d_in[4];     // [N]
  // d_in[3] = batch_id (implicit i/Lg), d_in[5] = topk scalar (compile-time KNN)

  // ---- params: jax pytree flattening (sorted dict keys) after 6 inputs ----
  auto P = [&](int i) { return (const float*)d_in[6 + i]; };
  // 0 chain_emb | 1 coord_pred.W 2 coord_pred.b
  // 3 ee.bn.b 4 ee.bn.g 5 ee.l1.W 6 ee.l1.b 7 ee.l2.W 8 ee.l2.b
  // layers: base 9 + 35*l (offsets in code below)
  // 114 ne.bn.b 115 ne.bn.g 116 ne.l1.W 117 ne.l1.b 118 ne.l2.W 119 ne.l2.b
  // 120 nie.W 121 nie.b

  // ---- workspace carve ----
  size_t off = 0;
  auto alloc = [&](size_t bytes) {
    char* p = (char*)d_ws + off;
    off += (bytes + 255) & ~(size_t)255;
    return (void*)p;
  };
  int*   dknn = (int*)  alloc((size_t)Eg * 4);
  float* cat  = (float*)alloc((size_t)Eg * 3 * H * 4);
  float* ta   = (float*)alloc((size_t)Eg * H * 4);
  float* tb   = (float*)alloc((size_t)Eg * H * 4);
  float* hE   = (float*)alloc((size_t)Eg * H * 4);
  float* wbuf = (float*)alloc((size_t)Eg * NHEADS * 4);
  float* att  = (float*)alloc((size_t)Eg * NHEADS * 4);
  float* hV   = (float*)alloc((size_t)Ng * H * 4);
  float* aggb = (float*)alloc((size_t)Ng * H * 4);
  float* t1   = (float*)alloc((size_t)Ng * 4 * H * 4);
  float* t2   = (float*)alloc((size_t)Ng * H * 4);
  float* t3   = (float*)alloc((size_t)Ng * H * 4);
  float* cv   = (float*)alloc((size_t)Bg * H * 4);
  float* cv2  = (float*)alloc((size_t)Bg * H * 4);
  float* mean = (float*)alloc(512 * 4);
  float* var  = (float*)alloc(512 * 4);

  const long nEH = (long)Eg * H, nNH = (long)Ng * H;

  // ---- kNN graph ----
  knn_kernel<<<(Ng + 255) / 256, 256, 0, stream>>>(Xin, dknn);

  // ---- node embedding: hV = hV0 + l2(relu(bn(l1(X.reshape(N,12))))) ----
  launch_gemm(Xin, P(116), P(117), t2, Ng, 12, H, 0, stream);
  bn_stats_kernel<<<H, 256, 0, stream>>>(t2, Ng, H, mean, var);
  bn_apply_kernel<<<ew_grid(nNH), 256, 0, stream>>>(t2, t2, mean, var, P(115), P(114), Ng, H, 1);
  launch_gemm(t2, P(118), P(119), t3, Ng, H, H, 0, stream);
  add2_kernel<<<ew_grid(nNH), 256, 0, stream>>>(hV, hV0, t3, nNH);

  // ---- edge embedding ----
  cat2_kernel<<<ew_grid((long)Eg * 2 * H), 256, 0, stream>>>(hV, dknn, cat);
  launch_gemm(cat, P(5), P(6), ta, Eg, 2 * H, H, 0, stream);
  bn_stats_kernel<<<H, 256, 0, stream>>>(ta, Eg, H, mean, var);
  bn_apply_kernel<<<ew_grid(nEH), 256, 0, stream>>>(ta, ta, mean, var, P(4), P(3), Eg, H, 1);
  launch_gemm(ta, P(7), P(8), hE, Eg, H, H, 0, stream);
  chain_add_kernel<<<ew_grid(nEH), 256, 0, stream>>>(hE, chain, dknn, P(0));
  rbf_kernel<<<ew_grid((long)Eg * NRBF), 256, 0, stream>>>(nidx, dknn, cat);
  launch_gemm(cat, P(120), P(121), ta, Eg, NRBF, H, 0, stream);
  add_kernel<<<ew_grid(nEH), 256, 0, stream>>>(hE, ta, nEH);

  // ---- GNN layers ----
  for (int l = 0; l < 3; ++l) {
    const int LB = 9 + 35 * l;
    // h_E = edge_mlp([hV_src, hV_dst])
    cat2_kernel<<<ew_grid((long)Eg * 2 * H), 256, 0, stream>>>(hV, dknn, cat);
    launch_gemm(cat, P(LB + 19), P(LB + 20), hE, Eg, 2 * H, H, 0, stream);
    // attention logits
    cat3_kernel<<<ew_grid((long)Eg * 3 * H), 256, 0, stream>>>(hV, hE, dknn, cat);
    launch_gemm(cat, P(LB + 3), P(LB + 4), ta, Eg, 3 * H, H, 1, stream);
    launch_gemm(ta, P(LB + 5), P(LB + 6), tb, Eg, H, H, 1, stream);
    launch_gemm(tb, P(LB + 7), P(LB + 8), wbuf, Eg, H, NHEADS, 0, stream);
    // V = gelu(WV h_E)
    launch_gemm(hE, P(LB + 1), P(LB + 2), ta, Eg, H, H, 2, stream);
    // softmax over contiguous 30-edge segments + aggregate
    segsoftmax_kernel<<<(Ng * NHEADS + 255) / 256, 256, 0, stream>>>(wbuf, att);
    agg_kernel<<<(Ng * H + 255) / 256, 256, 0, stream>>>(att, ta, aggb);
    // gated residual: hV += (agg @ WO) * sigmoid(gate(agg))
    launch_gemm(aggb, P(LB + 0), nullptr, t2, Ng, H, H, 0, stream);
    launch_gemm(aggb, P(LB + 9), P(LB + 10), t3, Ng, H, H, 0, stream);
    gated_res_kernel<<<ew_grid(nNH), 256, 0, stream>>>(hV, t2, t3, nNH);
    // node FFN with BN sandwich
    bn_stats_kernel<<<H, 256, 0, stream>>>(hV, Ng, H, mean, var);
    bn_apply_kernel<<<ew_grid(nNH), 256, 0, stream>>>(hV, t2, mean, var, P(LB + 22), P(LB + 21), Ng, H, 0);
    launch_gemm(t2, P(LB + 25), P(LB + 26), t1, Ng, H, 4 * H, 1, stream);
    launch_gemm(t1, P(LB + 27), P(LB + 28), t2, Ng, 4 * H, H, 0, stream);
    bn_stats_kernel<<<H, 256, 0, stream>>>(t2, Ng, H, mean, var);
    bn_apply_kernel<<<ew_grid(nNH), 256, 0, stream>>>(t2, t3, mean, var, P(LB + 24), P(LB + 23), Ng, H, 0);
    add_kernel<<<ew_grid(nNH), 256, 0, stream>>>(hV, t3, nNH);
    // per-graph context gate
    graph_mean_kernel<<<(Bg * H + 255) / 256, 256, 0, stream>>>(hV, cv);
    launch_gemm(cv, P(LB + 29), P(LB + 30), cv2, Bg, H, H, 1, stream);
    launch_gemm(cv2, P(LB + 31), P(LB + 32), cv, Bg, H, H, 1, stream);
    launch_gemm(cv, P(LB + 33), P(LB + 34), cv2, Bg, H, H, 0, stream);
    graph_gate_kernel<<<ew_grid(nNH), 256, 0, stream>>>(hV, cv2);
    // edge update
    cat3_kernel<<<ew_grid((long)Eg * 3 * H), 256, 0, stream>>>(hV, hE, dknn, cat);
    launch_gemm(cat, P(LB + 13), P(LB + 14), ta, Eg, 3 * H, H, 2, stream);
    launch_gemm(ta, P(LB + 15), P(LB + 16), tb, Eg, H, H, 2, stream);
    launch_gemm(tb, P(LB + 17), P(LB + 18), ta, Eg, H, H, 0, stream);
    add_kernel<<<ew_grid(nEH), 256, 0, stream>>>(hE, ta, nEH);
    bn_stats_kernel<<<H, 256, 0, stream>>>(hE, Eg, H, mean, var);
    bn_apply_kernel<<<ew_grid(nEH), 256, 0, stream>>>(hE, hE, mean, var, P(LB + 12), P(LB + 11), Eg, H, 0);
  }

  // ---- outputs: [X_out (N*12) | h_V (N*H)] ----
  float* out = (float*)d_out;
  launch_gemm(hV, P(1), P(2), out, Ng, H, 12, 0, stream);
  copy_kernel<<<ew_grid(nNH), 256, 0, stream>>>(out + (size_t)Ng * 12, hV, nNH);
}
